// Attention_13778255085616
// MI455X (gfx1250) — matrix operand order repaired and checked by reading, not compiled, not run on hardware
//
#include <hip/hip_runtime.h>

typedef __attribute__((ext_vector_type(16))) __bf16 v16bf;
typedef __attribute__((ext_vector_type(8)))  float  v8f;

#define HEADS 16
#define SEQN  512
#define DHEAD 64
#define LPAST 16384
#define LTOT  (LPAST + SEQN)   // 16896

__device__ __forceinline__ unsigned short f2bf(float f) {
    unsigned int u = __builtin_bit_cast(unsigned int, f);
    unsigned int r = (u + 0x7FFFu + ((u >> 16) & 1u)) >> 16;
    return (unsigned short)r;
}

struct BF16Frag { uint4 a, b; };
__device__ __forceinline__ v16bf make_frag(uint4 lo, uint4 hi) {
    BF16Frag u{lo, hi};
    return __builtin_bit_cast(v16bf, u);
}
// 16 bf16 elements = two 16-byte chunks; p0/p1 must be 16B-aligned
__device__ __forceinline__ v16bf ld_frag(const unsigned short* p0,
                                         const unsigned short* p1) {
    return make_frag(*(const uint4*)p0, *(const uint4*)p1);
}
__device__ __forceinline__ ushort4 f2bf4(float4 f) {
    ushort4 u; u.x = f2bf(f.x); u.y = f2bf(f.y); u.z = f2bf(f.z); u.w = f2bf(f.w);
    return u;
}

// LDS byte offset of a __shared__ object (generic AMDGPU pointer low bits)
__device__ __forceinline__ unsigned lds_off(const void* p) {
    return (unsigned)(unsigned long long)p;
}
// async 16B global -> LDS copy (per-lane addresses), tracked by ASYNCcnt
__device__ __forceinline__ void async_g2l_b128(unsigned ldsaddr, const float* gaddr) {
    asm volatile("global_load_async_to_lds_b128 %0, %1, off"
                 :: "v"(ldsaddr), "v"(gaddr) : "memory");
}

// ---------------------------------------------------------------------------
// bf16-WMMA GEMM:  out[M,N] = A[M,K] @ B[K,N] + bias[N]
// mode 0: N==3072 qkv output, scatter into q(f32+bf16)/k_new/v_new [H][N][D]
// mode 1: plain fp32 output
// block = 256 threads (8 waves), tile M=64 x N=128, K-step 32
// W tile stored TRANSPOSED in LDS so B-fragments are lane-contiguous.
// ---------------------------------------------------------------------------
__global__ __launch_bounds__(256)
void gemm_bf16_wmma(const float* __restrict__ A,
                    const float* __restrict__ Bw,
                    const float* __restrict__ bias,
                    int M, int N, int K, int mode,
                    float* __restrict__ out_plain,
                    float* __restrict__ qf, float* __restrict__ kf,
                    float* __restrict__ vf, unsigned short* __restrict__ qb)
{
    __shared__ unsigned short xs[64][32];     // A tile [m][k]
    __shared__ unsigned short wshT[128][32];  // B tile transposed [col][k]

    const int t    = threadIdx.x;
    const int lane = t & 31;
    const int w    = t >> 5;
    const int wm   = w >> 1;           // 0..3  (16-row band)
    const int wn   = w & 1;            // 0..1  (64-col band)
    const int bm   = blockIdx.y * 64;
    const int bn   = blockIdx.x * 128;

    v8f acc[4] = {};

    const int m   = lane & 15;
    const int kb  = (lane < 16) ? 0 : 8;    // A-frag K base (elements)
    const int kb2 = (lane < 16) ? 0 : 16;   // unused; B-frag also uses kb
    const int nl  = lane & 15;

    for (int k0 = 0; k0 < K; k0 += 32) {
        __syncthreads();
        // A tile: 64x32 = 512 float4 loads
        #pragma unroll
        for (int i = 0; i < 2; ++i) {
            int l4 = t + i * 256;               // 0..511
            int r = l4 >> 3, c4 = (l4 & 7) * 4;
            float4 f = *(const float4*)&A[(size_t)(bm + r) * K + k0 + c4];
            *(ushort4*)&xs[r][c4] = f2bf4(f);
        }
        // B tile: 32x128 = 1024 float4 loads, written transposed
        #pragma unroll
        for (int i = 0; i < 4; ++i) {
            int l4 = t + i * 256;               // 0..1023
            int r = l4 >> 5, c4 = (l4 & 31) * 4;
            float4 f = *(const float4*)&Bw[(size_t)(k0 + r) * N + bn + c4];
            wshT[c4 + 0][r] = f2bf(f.x);
            wshT[c4 + 1][r] = f2bf(f.y);
            wshT[c4 + 2][r] = f2bf(f.z);
            wshT[c4 + 3][r] = f2bf(f.w);
        }
        __syncthreads();

        v16bf af = ld_frag(&xs[wm * 16 + m][kb], &xs[wm * 16 + m][kb + 16]);
        #pragma unroll
        for (int ns = 0; ns < 4; ++ns) {
            const unsigned short* col = wshT[wn * 64 + ns * 16 + nl];
            v16bf bf = ld_frag(&col[kb], &col[kb + 16]);
            acc[ns] = __builtin_amdgcn_wmma_f32_16x16x32_bf16(
                false, af, false, bf, (short)0, acc[ns], false, false);
        }
    }

    // epilogue (C layout: lanes 0-15 rows 0-7, lanes 16-31 rows 8-15)
    const int rbase = (lane < 16) ? 0 : 8;
    #pragma unroll
    for (int ns = 0; ns < 4; ++ns) {
        #pragma unroll
        for (int r = 0; r < 8; ++r) {
            int row = bm + wm * 16 + rbase + r;
            int col = bn + wn * 64 + ns * 16 + nl;
            float v = acc[ns][r] + bias[col];
            if (mode == 1) {
                out_plain[(size_t)row * N + col] = v;
            } else {
                int s = col >> 10;            // 0=q 1=k 2=v
                int h = (col >> 6) & 15;
                int d = col & 63;
                size_t off = ((size_t)h * SEQN + row) * DHEAD + d;
                if (s == 0)      { qf[off] = v; qb[off] = f2bf(v); }
                else if (s == 1) { kf[off] = v; }
                else             { vf[off] = v; }
            }
        }
    }
}

// ---------------------------------------------------------------------------
// Flash attention over the concatenated cache.
// grid = (H, 8), block = 128 threads (4 waves); each wave owns 16 q rows.
// K stored natural [key][d] (B-frag of Q*K^T is lane-contiguous),
// V stored transposed [d][key] (B-frag of P*V is lane-contiguous).
// ---------------------------------------------------------------------------
__global__ __launch_bounds__(128)
void flash_attn(const float* __restrict__ past_k, const float* __restrict__ past_v,
                const float* __restrict__ knew,   const float* __restrict__ vnew,
                const unsigned short* __restrict__ qb,
                float* __restrict__ attn)
{
    __shared__ unsigned short Ks[32][DHEAD];   // K chunk [key][d]
    __shared__ unsigned short VsT[DHEAD][32];  // V chunk transposed [d][key]
    __shared__ unsigned short Pst[4][16][32];  // per-wave P staging [row][key]

    const int t    = threadIdx.x;
    const int lane = t & 31;
    const int w    = t >> 5;                   // 0..3
    const int h    = blockIdx.x;
    const int q0   = blockIdx.y * 64 + w * 16;

    const int m   = lane & 15;
    const int kb  = (lane < 16) ? 0 : 8;
    const int kb2 = (lane < 16) ? 0 : 16;
    const int nl  = lane & 15;
    const int rb  = (lane < 16) ? 0 : 8;

    // preload Q as two 16x32 A fragments (d = 0..31, 32..63), 128-bit loads
    const unsigned short* qrow = &qb[((size_t)h * SEQN + q0 + m) * DHEAD];
    v16bf aQ[2];
    #pragma unroll
    for (int dc = 0; dc < 2; ++dc)
        aQ[dc] = ld_frag(&qrow[dc * 32 + kb], &qrow[dc * 32 + kb + 16]);

    v8f accO[4] = {};
    float mrow[8], lrow[8];
    #pragma unroll
    for (int r = 0; r < 8; ++r) { mrow[r] = -3.0e38f; lrow[r] = 0.f; }
    const float scale = 0.125f;   // 1/sqrt(64)

    for (int c0 = 0; c0 < LTOT; c0 += 32) {
        __syncthreads();
        // cooperative K/V chunk load: 512 float4 per array, 128 threads
        #pragma unroll
        for (int i = 0; i < 4; ++i) {
            int l4 = t + i * 128;               // 0..511
            int key = l4 >> 4, d4 = (l4 & 15) * 4;
            int l = c0 + key;
            const float *kp, *vp;
            if (l < LPAST) {
                kp = &past_k[((size_t)h * LPAST + l) * DHEAD];
                vp = &past_v[((size_t)h * LPAST + l) * DHEAD];
            } else {
                kp = &knew[((size_t)h * SEQN + (l - LPAST)) * DHEAD];
                vp = &vnew[((size_t)h * SEQN + (l - LPAST)) * DHEAD];
            }
            float4 fk = *(const float4*)&kp[d4];
            float4 fv = *(const float4*)&vp[d4];
            *(ushort4*)&Ks[key][d4] = f2bf4(fk);
            VsT[d4 + 0][key] = f2bf(fv.x);
            VsT[d4 + 1][key] = f2bf(fv.y);
            VsT[d4 + 2][key] = f2bf(fv.z);
            VsT[d4 + 3][key] = f2bf(fv.w);
        }
        __syncthreads();

        // prefetch next chunk's K rows (one 256B row per lane of wave 0)
        if (t < 32 && c0 + 32 < LTOT) {
            int l = c0 + 32 + t;
            const float* pp = (l < LPAST)
                ? &past_k[((size_t)h * LPAST + l) * DHEAD]
                : &knew[((size_t)h * SEQN + (l - LPAST)) * DHEAD];
            __builtin_prefetch(pp, 0, 1);
        }

        // S = Q K^T : two 16x16 score tiles (2 WMMA each over d)
        v8f S[2];
        #pragma unroll
        for (int ks = 0; ks < 2; ++ks) {
            v8f s = {};
            #pragma unroll
            for (int dc = 0; dc < 2; ++dc) {
                const unsigned short* krow = Ks[ks * 16 + nl];
                v16bf bk = ld_frag(&krow[dc * 32 + kb], &krow[dc * 32 + kb + 16]);
                s = __builtin_amdgcn_wmma_f32_16x16x32_bf16(
                    false, aQ[dc], false, bk, (short)0, s, false, false);
            }
            S[ks] = s;
        }

        // fp32 online softmax (per row; reduce over the 16 lanes of a half)
        float p0[8], p1[8];
        #pragma unroll
        for (int r = 0; r < 8; ++r) {
            float s0 = S[0][r] * scale, s1 = S[1][r] * scale;
            float tm = fmaxf(s0, s1);
            #pragma unroll
            for (int off = 8; off >= 1; off >>= 1)
                tm = fmaxf(tm, __shfl_xor(tm, off, 32));
            float mn = fmaxf(mrow[r], tm);
            float alpha = __expf(mrow[r] - mn);
            float e0 = __expf(s0 - mn), e1 = __expf(s1 - mn);
            float rs = e0 + e1;
            #pragma unroll
            for (int off = 8; off >= 1; off >>= 1)
                rs += __shfl_xor(rs, off, 32);
            lrow[r] = lrow[r] * alpha + rs;
            mrow[r] = mn;
            #pragma unroll
            for (int ns = 0; ns < 4; ++ns) accO[ns][r] *= alpha;
            p0[r] = e0; p1[r] = e1;
        }

        // stage P (C layout) -> LDS -> reload as 16x32 A fragment
        #pragma unroll
        for (int r = 0; r < 8; ++r) {
            Pst[w][rb + r][nl]      = f2bf(p0[r]);
            Pst[w][rb + r][16 + nl] = f2bf(p1[r]);
        }
        v16bf aP = ld_frag(&Pst[w][m][kb], &Pst[w][m][kb + 16]);
        // O += P V
        #pragma unroll
        for (int ns = 0; ns < 4; ++ns) {
            const unsigned short* vrow = VsT[ns * 16 + nl];
            v16bf bv = ld_frag(&vrow[kb], &vrow[kb + 16]);
            accO[ns] = __builtin_amdgcn_wmma_f32_16x16x32_bf16(
                false, aP, false, bv, (short)0, accO[ns], false, false);
        }
    }

    // normalize and write attn[row][h*64 + d]
    #pragma unroll
    for (int ns = 0; ns < 4; ++ns) {
        #pragma unroll
        for (int r = 0; r < 8; ++r) {
            int row = q0 + rb + r;
            int d   = ns * 16 + nl;
            attn[(size_t)row * 1024 + h * DHEAD + d] = accO[ns][r] / lrow[r];
        }
    }
}

// ---------------------------------------------------------------------------
// Probe scoring: full fp32 online-softmax attention for 32 probed queries.
// grid = (H), block = 256 (8 waves); wave w handles probes 4w..4w+3.
// K/V streamed through LDS with double-buffered ASYNC global->LDS copies.
// ---------------------------------------------------------------------------
__global__ __launch_bounds__(256)
void probe_kernel(const float* __restrict__ past_k, const float* __restrict__ past_v,
                  const float* __restrict__ knew,   const float* __restrict__ vnew,
                  const float* __restrict__ qf, const int* __restrict__ probe_idx,
                  float* __restrict__ oprobe)
{
    __shared__ float Kst[2][64][DHEAD];   // [buf][key][d], 16 KB per buf
    __shared__ float Vst[2][64][DHEAD];

    const int h    = blockIdx.x;
    const int t    = threadIdx.x;
    const int lane = t & 31;
    const int w    = t >> 5;
    const int d0   = lane * 2;
    const float scale = 0.125f;

    float q0[4], q1[4], m_[4], l_[4], a0[4], a1[4];
    #pragma unroll
    for (int i = 0; i < 4; ++i) {
        int p = w * 4 + i;
        int row = probe_idx[p];
        q0[i] = qf[((size_t)h * SEQN + row) * DHEAD + d0];
        q1[i] = qf[((size_t)h * SEQN + row) * DHEAD + d0 + 1];
        m_[i] = -3.0e38f; l_[i] = 0.f; a0[i] = 0.f; a1[i] = 0.f;
    }

    // issue async copies of one 64-key chunk (each wave: 4 K + 4 V = 8 ops)
    auto stage = [&](int c, int buf) {
        #pragma unroll
        for (int i = 0; i < 4; ++i) {
            int l4 = t + i * 256;               // 0..1023
            int key = l4 >> 4, d4 = (l4 & 15) * 4;
            int l = c * 64 + key;
            const float *kp, *vp;
            if (l < LPAST) {
                kp = &past_k[((size_t)h * LPAST + l) * DHEAD + d4];
                vp = &past_v[((size_t)h * LPAST + l) * DHEAD + d4];
            } else {
                kp = &knew[((size_t)h * SEQN + (l - LPAST)) * DHEAD + d4];
                vp = &vnew[((size_t)h * SEQN + (l - LPAST)) * DHEAD + d4];
            }
            async_g2l_b128(lds_off(&Kst[buf][key][d4]), kp);
            async_g2l_b128(lds_off(&Vst[buf][key][d4]), vp);
        }
    };

    const int NCH = LTOT / 64;   // 264
    stage(0, 0);
    for (int c = 0; c < NCH; ++c) {
        int buf = c & 1;
        if (c + 1 < NCH) {
            stage(c + 1, buf ^ 1);
            asm volatile("s_wait_asynccnt 0x8" ::: "memory");  // chunk c done
        } else {
            asm volatile("s_wait_asynccnt 0x0" ::: "memory");
        }
        __syncthreads();   // all waves' slices visible

        for (int kk = 0; kk < 64; ++kk) {
            float2 kv = *(const float2*)&Kst[buf][kk][d0];
            float2 vv = *(const float2*)&Vst[buf][kk][d0];
            #pragma unroll
            for (int i = 0; i < 4; ++i) {
                float s = q0[i] * kv.x + q1[i] * kv.y;
                #pragma unroll
                for (int off = 16; off >= 1; off >>= 1)
                    s += __shfl_xor(s, off, 32);
                s *= scale;
                float mn = fmaxf(m_[i], s);
                float al = __expf(m_[i] - mn);
                float e  = __expf(s - mn);
                a0[i] = a0[i] * al + e * vv.x;
                a1[i] = a1[i] * al + e * vv.y;
                l_[i] = l_[i] * al + e;
                m_[i] = mn;
            }
        }
        __syncthreads();   // reads done before buf is overwritten again
    }

    #pragma unroll
    for (int i = 0; i < 4; ++i) {
        int p = w * 4 + i;
        oprobe[((size_t)h * 32 + p) * DHEAD + d0]     = a0[i] / l_[i];
        oprobe[((size_t)h * 32 + p) * DHEAD + d0 + 1] = a1[i] / l_[i];
    }
}

// ---------------------------------------------------------------------------
extern "C" void kernel_launch(void* const* d_in, const int* in_sizes, int n_in,
                              void* d_out, int out_size, void* d_ws, size_t ws_size,
                              hipStream_t stream) {
    const float* x        = (const float*)d_in[0];
    const float* past_k   = (const float*)d_in[1];
    const float* past_v   = (const float*)d_in[2];
    const float* W_qkv    = (const float*)d_in[3];
    const float* b_qkv    = (const float*)d_in[4];
    const float* W_proj   = (const float*)d_in[5];
    const float* b_proj   = (const float*)d_in[6];
    const int*   probe_ix = (const int*)d_in[7];

    float* out    = (float*)d_out;                 // [512,1024]
    float* oprobe = out + (size_t)SEQN * 1024;     // [16,32,64]

    char* ws = (char*)d_ws;
    float*          qf   = (float*)(ws);                              // 2 MB
    float*          kf   = (float*)(ws + ((size_t)2 << 20));          // 2 MB
    float*          vf   = (float*)(ws + ((size_t)4 << 20));          // 2 MB
    float*          attn = (float*)(ws + ((size_t)6 << 20));          // 2 MB
    unsigned short* qb   = (unsigned short*)(ws + ((size_t)8 << 20)); // 1 MB

    // 1) QKV projection (512x1024 @ 1024x3072), scatter into q/k/v
    gemm_bf16_wmma<<<dim3(24, 8), dim3(256), 0, stream>>>(
        x, W_qkv, b_qkv, SEQN, 3072, 1024, 0, nullptr, qf, kf, vf, qb);
    // 2) Flash attention over [past | new] cache
    flash_attn<<<dim3(HEADS, 8), dim3(128), 0, stream>>>(
        past_k, past_v, kf, vf, qb, attn);
    // 3) Output projection (512x1024 @ 1024x1024) -> d_out
    gemm_bf16_wmma<<<dim3(8, 8), dim3(256), 0, stream>>>(
        attn, W_proj, b_proj, SEQN, 1024, 1024, 1, out,
        nullptr, nullptr, nullptr, nullptr);
    // 4) fp32 probe attention (async-LDS pipelined) -> d_out tail
    probe_kernel<<<dim3(HEADS), dim3(256), 0, stream>>>(
        past_k, past_v, kf, vf, qf, probe_ix, oprobe);
}
